// CTCLossLayer_17781164605942
// MI455X (gfx1250) — compile-verified, matrix operand order checked by reading, not run on hardware
//
#include <hip/hip_runtime.h>
#include <hip/hip_bf16.h>
#include <stdint.h>

typedef __attribute__((ext_vector_type(16))) _Float16 v16h;
typedef __attribute__((ext_vector_type(8)))  float    v8f;

#define CDIM 128
#define NEGV (-1e30f)
#define MAXS 160

// ---------------------------------------------------------------------------
// Kernel A: per-(b,t) log-sum-exp normalizer (via WMMA row-sum) + argmax.
// Block = 256 threads (8 waves). Each wave handles 16 timesteps of one batch b.
// ---------------------------------------------------------------------------
__global__ void lse_argmax_kernel(const float* __restrict__ logits,
                                  float* __restrict__ lse,
                                  int* __restrict__ pred,
                                  int T) {
    const int b   = blockIdx.x;
    const int t0  = blockIdx.y * 128;
    const int tid = threadIdx.x;
    const int w   = tid >> 5;          // wave id 0..7
    const int l   = tid & 31;          // lane
    const int g   = l >> 4;            // lane group (0/1)
    const int m   = l & 15;            // matrix row within group

    __shared__ _Float16 tiles[8][16 * CDIM];   // exp values, f16
    __shared__ float    rmax [8][16];          // per-row max

    #pragma unroll 4
    for (int r = 0; r < 16; ++r) {
        const int t = t0 + w * 16 + r;
        _Float16* dst = &tiles[w][r * CDIM + l * 4];
        if (t < T) {
            const float4 v4 = reinterpret_cast<const float4*>(
                logits + ((size_t)b * T + t) * CDIM)[l];
            // local argmax over the 4 elements (first occurrence wins)
            float mv = v4.x; int mi = l * 4;
            if (v4.y > mv) { mv = v4.y; mi = l * 4 + 1; }
            if (v4.z > mv) { mv = v4.z; mi = l * 4 + 2; }
            if (v4.w > mv) { mv = v4.w; mi = l * 4 + 3; }
            #pragma unroll
            for (int off = 16; off > 0; off >>= 1) {
                float ov = __shfl_xor(mv, off, 32);
                int   oi = __shfl_xor(mi, off, 32);
                if (ov > mv || (ov == mv && oi < mi)) { mv = ov; mi = oi; }
            }
            if (l == 0) {
                pred[(size_t)b * T + t] = mi;
                rmax[w][r] = mv;
            }
            dst[0] = (_Float16)__expf(v4.x - mv);
            dst[1] = (_Float16)__expf(v4.y - mv);
            dst[2] = (_Float16)__expf(v4.z - mv);
            dst[3] = (_Float16)__expf(v4.w - mv);
        } else {
            dst[0] = (_Float16)0.f; dst[1] = (_Float16)0.f;
            dst[2] = (_Float16)0.f; dst[3] = (_Float16)0.f;
        }
    }
    __syncthreads();

    // B = all-ones 32x16 f16 matrix (layout-invariant since uniform).
    v16h ones;
    #pragma unroll
    for (int i = 0; i < 16; ++i) ones[i] = (_Float16)1.0f;

    v8f acc = {};
    #pragma unroll
    for (int kc = 0; kc < 4; ++kc) {
        v16h a;
        #pragma unroll
        for (int e = 0; e < 16; ++e) {
            const int v = e >> 1, h = e & 1;
            const int koff = kc * 32 + ((v >= 4) ? 16 : 0) + g * 8 + (v & 3) * 2 + h;
            a[e] = tiles[w][m * CDIM + koff];
        }
        acc = __builtin_amdgcn_wmma_f32_16x16x32_f16(
            false, a, false, ones, (short)0, acc, false, false);
    }

    // D[M = r + 8g][N = m] = rowsum(exp) for timestep row M, replicated over N.
    if (m == 0) {
        #pragma unroll
        for (int r2 = 0; r2 < 8; ++r2) {
            const int M = r2 + g * 8;
            const int t = t0 + w * 16 + M;
            if (t < T)
                lse[(size_t)b * T + t] = rmax[w][M] + __logf(acc[r2]);
        }
    }
}

// ---------------------------------------------------------------------------
// Kernel B: CTC alpha recursion. One wave32 per batch row.
// Raw-logit recursion; subtract sum(lse) at the end. Depth-4 async LDS pipeline.
// ---------------------------------------------------------------------------
__global__ void ctc_alpha_kernel(const float* __restrict__ logits,
                                 const float* __restrict__ lse,
                                 const int* __restrict__ labels,
                                 const int* __restrict__ label_len,
                                 const int* __restrict__ logit_len,
                                 float* __restrict__ loss_b,
                                 int T, int L, int C) {
    const int b    = blockIdx.x;
    const int lane = threadIdx.x;
    const int S    = 2 * L + 1;
    const int blank = C - 1;
    const int ll   = label_len[b];
    const int len  = logit_len[b];

    __shared__ float rowbuf[4][CDIM];
    __shared__ float Ab[2][MAXS + 2];
    __shared__ int   exts[MAXS];
    __shared__ int   allow[MAXS];

    for (int s = lane; s < S; s += 32)
        exts[s] = (s & 1) ? labels[(size_t)b * L + (s >> 1)] : blank;
    __syncthreads();
    for (int s = lane; s < S; s += 32)
        allow[s] = (s >= 2 && (s & 1) && exts[s] != exts[s - 2]) ? 1 : 0;
    if (lane < 2) { Ab[0][lane] = NEGV; Ab[1][lane] = NEGV; }

    // sum of per-step log-normalizers (produced by kernel A)
    float sl = 0.f;
    for (int t = lane; t < len; t += 32) sl += lse[(size_t)b * T + t];
    #pragma unroll
    for (int off = 16; off > 0; off >>= 1) sl += __shfl_xor(sl, off, 32);
    __syncthreads();

    const size_t rowbase = (size_t)b * T;
    // preload rows 0..2 (clamped duplicates are harmless)
    #pragma unroll
    for (int d = 0; d < 3; ++d) {
        const int tt = (d < len) ? d : (len - 1);
        unsigned long long ga = (unsigned long long)(uintptr_t)
            (logits + (rowbase + tt) * CDIM) + (unsigned)(lane * 16);
        unsigned int la = (unsigned int)(uintptr_t)(&rowbuf[d][0]) + (unsigned)(lane * 16);
        asm volatile("global_load_async_to_lds_b128 %0, %1, off"
                     :: "v"(la), "v"(ga) : "memory");
    }

    int cur = 0, nxt = 1;
    for (int t = 0; t < len; ++t) {
        {   // prefetch row t+3 into slot (t+3)&3 (clamped at the tail)
            const int tp = t + 3;
            const int tt = (tp < len) ? tp : (len - 1);
            unsigned long long ga = (unsigned long long)(uintptr_t)
                (logits + (rowbase + tt) * CDIM) + (unsigned)(lane * 16);
            unsigned int la = (unsigned int)(uintptr_t)(&rowbuf[tp & 3][0]) + (unsigned)(lane * 16);
            asm volatile("global_load_async_to_lds_b128 %0, %1, off"
                         :: "v"(la), "v"(ga) : "memory");
        }
        asm volatile("s_wait_asynccnt 0x3" ::: "memory");

        float* Ap = Ab[cur];
        float* Ac = Ab[nxt];
        const int slot = t & 3;
        if (t == 0) {
            #pragma unroll
            for (int j = 0; j < 5; ++j) {
                const int s = lane + 32 * j;
                if (s < S) Ac[2 + s] = (s < 2) ? rowbuf[0][exts[s]] : NEGV;
            }
        } else {
            #pragma unroll
            for (int j = 0; j < 5; ++j) {
                const int s = lane + 32 * j;
                if (s < S) {
                    const float a1 = Ap[2 + s];
                    const float a2 = Ap[1 + s];
                    const float a3 = allow[s] ? Ap[s] : NEGV;
                    float mm  = fmaxf(a1, fmaxf(a2, a3));
                    float sum = __expf(a1 - mm) + __expf(a2 - mm) + __expf(a3 - mm);
                    Ac[2 + s] = mm + __logf(sum) + rowbuf[slot][exts[s]];
                }
            }
        }
        cur ^= 1; nxt ^= 1;
    }

    if (lane == 0) {
        float* Af = Ab[cur];                    // last-written buffer
        const float e1 = Af[2 + 2 * ll];
        const float e2 = Af[2 + 2 * ll - 1];
        const float mm = fmaxf(e1, e2);
        loss_b[b] = sl - (mm + __logf(__expf(e1 - mm) + __expf(e2 - mm)));
    }
}

// ---------------------------------------------------------------------------
// Kernel C: greedy decode (collapse) fused with Levenshtein DP.
// One wave32 per batch row; DP row (L+1 = 65 cells) lives in registers,
// insertion handled by the jr + cummin(cand - jr) trick via a wave min-scan.
// ---------------------------------------------------------------------------
__global__ void decode_edit_kernel(const int* __restrict__ pred,
                                   const int* __restrict__ labels,
                                   const int* __restrict__ label_len,
                                   const int* __restrict__ logit_len,
                                   float* __restrict__ ler_b,
                                   int T, int L, int C) {
    const int b    = blockIdx.x;
    const int lane = threadIdx.x;
    const int blank = C - 1;
    const int len  = logit_len[b];
    const int j0   = lane * 3;
    const int BIG  = 0x3fffffff;

    int rowreg[3], labreg[3];
    #pragma unroll
    for (int k = 0; k < 3; ++k) {
        const int j = j0 + k;
        rowreg[k] = j;
        labreg[k] = (j >= 1 && j <= L) ? labels[(size_t)b * L + j - 1] : -1234567;
    }

    int prevp = -1;
    for (int tb = 0; tb < T; tb += 32) {
        const int t_l = tb + lane;
        const int pv  = (t_l < T) ? pred[(size_t)b * T + t_l] : blank;
        const int ni  = (T - tb) < 32 ? (T - tb) : 32;
        for (int i = 0; i < ni; ++i) {
            const int p = __shfl(pv, i, 32);
            const int t = tb + i;
            const bool keep = (p != blank) && (p != prevp) && (t < len);
            prevp = p;
            if (keep) {
                const int c  = p;
                const int up = __shfl_up(rowreg[2], 1, 32);  // row[j0-1]
                int v[3];
                #pragma unroll
                for (int k = 0; k < 3; ++k) {
                    const int j = j0 + k;
                    if (j == 0) {
                        v[k] = rowreg[0] + 1;                 // cand[0] - 0
                    } else {
                        const int rm1 = (k == 0) ? up : rowreg[k - 1];
                        const int sub = rm1 + ((c != labreg[k]) ? 1 : 0);
                        const int del = rowreg[k] + 1;
                        v[k] = ((sub < del) ? sub : del) - j;  // cand[j] - j
                    }
                    if (j > L) v[k] = BIG;
                }
                v[1] = min(v[1], v[0]);
                v[2] = min(v[2], v[1]);
                int r = v[2];
                #pragma unroll
                for (int off = 1; off < 32; off <<= 1) {
                    const int o = __shfl_up(r, off, 32);
                    if (lane >= off) r = min(r, o);
                }
                int excl = __shfl_up(r, 1, 32);
                if (lane == 0) excl = BIG;
                #pragma unroll
                for (int k = 0; k < 3; ++k) {
                    const int j = j0 + k;
                    const int nv = min(v[k], excl) + j;       // jr + cummin
                    if (j <= L) rowreg[k] = nv;
                }
            }
        }
    }

    const int ll = label_len[b];
    const int kq = ll % 3, tq = ll / 3;
    const int val = (kq == 0) ? rowreg[0] : (kq == 1) ? rowreg[1] : rowreg[2];
    const int dist = __shfl(val, tq, 32);
    if (lane == 0) ler_b[b] = (float)dist / (float)ll;
}

// ---------------------------------------------------------------------------
// Kernel D: deterministic reduction of per-batch loss / LER into the 2 outputs.
// ---------------------------------------------------------------------------
__global__ void finalize_kernel(const float* __restrict__ loss_b,
                                const float* __restrict__ ler_b,
                                float* __restrict__ out, int B) {
    __shared__ float s1[256], s2[256];
    const int tid = threadIdx.x;
    float a = 0.f, c = 0.f;
    for (int i = tid; i < B; i += 256) { a += loss_b[i]; c += ler_b[i]; }
    s1[tid] = a; s2[tid] = c;
    __syncthreads();
    for (int off = 128; off > 0; off >>= 1) {
        if (tid < off) { s1[tid] += s1[tid + off]; s2[tid] += s2[tid + off]; }
        __syncthreads();
    }
    if (tid == 0) {
        out[0] = s1[0] / (float)B;   // mean CTC loss
        out[1] = s2[0] / (float)B;   // mean LER
    }
}

// ---------------------------------------------------------------------------
extern "C" void kernel_launch(void* const* d_in, const int* in_sizes, int n_in,
                              void* d_out, int out_size, void* d_ws, size_t ws_size,
                              hipStream_t stream) {
    const int*   labels    = (const int*)d_in[0];
    const float* logits    = (const float*)d_in[1];
    const int*   label_len = (const int*)d_in[2];
    const int*   logit_len = (const int*)d_in[3];

    const int B = in_sizes[2];
    const int L = in_sizes[0] / B;
    const int C = CDIM;                      // reference setup: C = 128
    const int T = in_sizes[1] / (B * C);

    float* out    = (float*)d_out;
    float* lse    = (float*)d_ws;
    int*   pred   = (int*)((char*)d_ws + (size_t)B * T * sizeof(float));
    float* loss_b = (float*)((char*)d_ws + (size_t)B * T * (sizeof(float) + sizeof(int)));
    float* ler_b  = loss_b + B;

    dim3 gA(B, (T + 127) / 128);
    lse_argmax_kernel<<<gA, 256, 0, stream>>>(logits, lse, pred, T);
    ctc_alpha_kernel<<<B, 32, 0, stream>>>(logits, lse, labels, label_len,
                                           logit_len, loss_b, T, L, C);
    decode_edit_kernel<<<B, 32, 0, stream>>>(pred, labels, label_len,
                                             logit_len, ler_b, T, L, C);
    finalize_kernel<<<1, 256, 0, stream>>>(loss_b, ler_b, out, B);
}
